// MultiHeadAttention_58832462020679
// MI455X (gfx1250) — compile-verified, matrix operand order checked by reading, not run on hardware
//
#include <hip/hip_runtime.h>
#include <math.h>
#include <stdint.h>

typedef __attribute__((ext_vector_type(16))) _Float16 v16h;
typedef __attribute__((ext_vector_type(8)))  _Float16 v8h;
typedef __attribute__((ext_vector_type(8)))  float    v8f;
typedef __attribute__((ext_vector_type(4)))  float    v4f;
typedef __attribute__((ext_vector_type(4)))  int      v4i;

constexpr int DM = 1024;   // d_model
constexpr int NH = 16;     // heads
constexpr int DK = 64;     // head dim
constexpr int BB = 2;      // batch
constexpr int SS = 2048;   // seq len

constexpr int BK  = 32;        // K depth per WMMA step (f16)
constexpr int PAD = 8;         // LDS row pad (80B stride -> conflict-free b128 reads)
constexpr int LDT = BK + PAD;  // 40 halves per LDS tile row

enum { MODE_F16 = 0, MODE_F32 = 1, MODE_VT = 2, MODE_SCORES = 3, MODE_PV = 4 };

// ---- CDNA5 async global->LDS copy (ASYNCcnt-tracked), with sync fallback ----
#if defined(__gfx1250__) && __has_builtin(__builtin_amdgcn_global_load_async_to_lds_b128)
#define ASYNC_LDS 1
#else
#define ASYNC_LDS 0
#endif

// Builtin signature is (int4 addrspace(1)*, int4 addrspace(3)*, imm offset, imm cpol).
typedef __attribute__((address_space(1))) v4i as1_v4i;
typedef __attribute__((address_space(3))) v4i as3_v4i;

__device__ __forceinline__ void copy16ToLds(const _Float16* g, _Float16* l) {
#if ASYNC_LDS
  // Generic LDS address keeps the byte offset in addr[31:0] (ISA 10.2 aperture map).
  __builtin_amdgcn_global_load_async_to_lds_b128(
      (as1_v4i*)(uintptr_t)g,
      (as3_v4i*)(uint32_t)(uintptr_t)l,
      0, 0);
#else
  *(v8h*)l = *(const v8h*)g;
#endif
}

template <int N>
__device__ __forceinline__ void waitAsync() {
#if ASYNC_LDS
#if __has_builtin(__builtin_amdgcn_s_wait_asynccnt)
  __builtin_amdgcn_s_wait_asynccnt(N);
#else
  asm volatile("s_wait_asynccnt %0" :: "n"(N) : "memory");
#endif
#endif
}

// Load a 16x32 f16 WMMA fragment (A or B, identical gather) from an LDS tile.
// Lane l: row = caller-provided (tileRow + (l&15)); halves 0..7 at K = (l>>4)*8,
// halves 8..15 at K = 16 + (l>>4)*8  (ISA 7.12.2, 16-bit A 16x32 layout).
__device__ __forceinline__ v16h ldsFrag(const _Float16* tile, int row, int lane) {
  const _Float16* p = tile + row * LDT + ((lane >> 4) << 3);
  v8h lo = *(const v8h*)p;
  v8h hi = *(const v8h*)(p + 16);
  v16h r;
#pragma unroll
  for (int i = 0; i < 8; ++i) { r[i] = lo[i]; r[i + 8] = hi[i]; }
  return r;
}

// C[M x N] = A[M x K] * B[N x K]^T  (A,B row-major; f32 accumulate via WMMA).
// 256 threads = 8 waves; double-buffered LDS tiles with async copy-ahead.
template <int BM, int BN, int MODE>
__global__ __launch_bounds__(256) void gemm_wmma(
    const void* __restrict__ Aall, int lda,
    const _Float16* __restrict__ Ball, int ldb,
    void* __restrict__ Call, int ldc,
    int Kd, float scale)
{
  constexpr int WROWS = BM / 32;       // waves along M
  constexpr int WCOLS = 8 / WROWS;     // waves along N
  constexpr int WNW   = BN / WCOLS;    // cols per wave
  constexpr int NT    = WNW / 16;      // 16-wide N tiles per wave
  constexpr int MT    = 2;             // 16-tall M tiles per wave
  constexpr int ACH   = (BM * (BK / 8)) / 256;  // A b128 copies per thread per tile
  constexpr int BCH   = (BN * (BK / 8)) / 256;  // B b128 copies per thread per tile
  // Async instructions issued per wave per tile (A staged via VGPRs in PV mode):
  constexpr int ISSUES = (MODE == MODE_PV) ? BCH : (ACH + BCH);

  __shared__ alignas(16) _Float16 As[2][BM * LDT];
  __shared__ alignas(16) _Float16 Bs[2][BN * LDT];

  const int tid  = threadIdx.x;
  const int lane = tid & 31;
  const int wave = tid >> 5;
  const int wm = wave % WROWS, wn = wave / WROWS;
  const int rowBase = wm * 32;
  const int colBase = wn * WNW;

  const int mBlk = blockIdx.y * BM;
  const int nBlk = blockIdx.x * BN;
  const int z  = blockIdx.z;
  const int bb = z >> 4, hh = z & 15;

  const _Float16* A16 = nullptr;
  const float*    A32 = nullptr;
  const _Float16* Bp  = Ball;
  if constexpr (MODE == MODE_SCORES) {
    A16 = (const _Float16*)Aall + (size_t)bb * SS * DM + hh * DK;  // Q head slice
    Bp  = Ball               + (size_t)bb * SS * DM + hh * DK;     // K head slice
  } else if constexpr (MODE == MODE_PV) {
    A32 = (const float*)Aall + (size_t)z * SS * SS;                // probs[b,h]
    Bp  = Ball               + (size_t)z * DK * SS;                // Vt[b,h]
  } else {
    A16 = (const _Float16*)Aall;
  }

  if constexpr (MODE == MODE_SCORES) {
    // Entire block above the causal diagonal: just emit the mask value.
    if (nBlk > mBlk + BM - 1) {
      float* C = (float*)Call + (size_t)z * SS * SS;
#pragma unroll
      for (int mt = 0; mt < MT; ++mt)
#pragma unroll
        for (int nt = 0; nt < NT; ++nt) {
          int n = nBlk + colBase + nt * 16 + (lane & 15);
#pragma unroll
          for (int j = 0; j < 8; ++j) {
            int m = mBlk + rowBase + mt * 16 + ((lane >> 4) << 3) + j;
            C[(size_t)m * SS + n] = -1e30f;
          }
        }
      return;
    }
  }

  // Stage one BK-deep tile pair into LDS buffer `buf` (async where possible).
  auto stage = [&](int k0, int buf) {
    if constexpr (MODE == MODE_PV) {
      // fp32 probs -> f16 conversion must round-trip through VGPRs.
#pragma unroll
      for (int it = 0; it < ACH; ++it) {
        int idx = tid + it * 256;
        int r = idx >> 2, c = (idx & 3) << 3;
        const float* src = A32 + (size_t)(mBlk + r) * lda + k0 + c;
        v4f f0 = *(const v4f*)src;
        v4f f1 = *(const v4f*)(src + 4);
        _Float16* d = &As[buf][r * LDT + c];
#pragma unroll
        for (int i = 0; i < 4; ++i) { d[i] = (_Float16)f0[i]; d[4 + i] = (_Float16)f1[i]; }
      }
    } else {
#pragma unroll
      for (int it = 0; it < ACH; ++it) {
        int idx = tid + it * 256;
        int r = idx >> 2, c = (idx & 3) << 3;
        copy16ToLds(A16 + (size_t)(mBlk + r) * lda + k0 + c, &As[buf][r * LDT + c]);
      }
    }
#pragma unroll
    for (int it = 0; it < BCH; ++it) {
      int idx = tid + it * 256;
      int r = idx >> 2, c = (idx & 3) << 3;
      copy16ToLds(Bp + (size_t)(nBlk + r) * ldb + k0 + c, &Bs[buf][r * LDT + c]);
    }
  };

  v8f acc[MT][NT] = {};
  const int nk = Kd / BK;

  stage(0, 0);
  for (int kk = 0; kk < nk; ++kk) {
    const int cur = kk & 1;
    if (kk + 1 < nk) {
      stage((kk + 1) * BK, cur ^ 1);   // copy-ahead: next tile in flight during compute
      waitAsync<ISSUES>();             // require only the current tile complete
    } else {
      waitAsync<0>();
    }
    __syncthreads();                   // current tile visible to all waves

    v16h af[MT], bf[NT];
#pragma unroll
    for (int mt = 0; mt < MT; ++mt)
      af[mt] = ldsFrag(As[cur], rowBase + mt * 16 + (lane & 15), lane);
#pragma unroll
    for (int nt = 0; nt < NT; ++nt)
      bf[nt] = ldsFrag(Bs[cur], colBase + nt * 16 + (lane & 15), lane);
#pragma unroll
    for (int mt = 0; mt < MT; ++mt)
#pragma unroll
      for (int nt = 0; nt < NT; ++nt)
        acc[mt][nt] = __builtin_amdgcn_wmma_f32_16x16x32_f16(
            false, af[mt], false, bf[nt], (short)0, acc[mt][nt], false, false);
    __syncthreads();                   // all waves done reading buf before reuse
  }

  // ---- store (C/D layout: lane l -> col l&15; VGPR j -> row j + (l>>4)*8) ----
#pragma unroll
  for (int mt = 0; mt < MT; ++mt)
#pragma unroll
    for (int nt = 0; nt < NT; ++nt) {
      int nl    = colBase + nt * 16 + (lane & 15);
      int mbase = rowBase + mt * 16 + ((lane >> 4) << 3);
#pragma unroll
      for (int j = 0; j < 8; ++j) {
        int m = mBlk + mbase + j;
        int n = nBlk + nl;
        float v = acc[mt][nt][j];
        if constexpr (MODE == MODE_F16) {
          ((_Float16*)Call)[(size_t)m * ldc + n] = (_Float16)v;
        } else if constexpr (MODE == MODE_F32) {
          ((float*)Call)[(size_t)m * ldc + n] = v;
        } else if constexpr (MODE == MODE_VT) {
          int b = m >> 11, s = m & (SS - 1);
          int h = n >> 6,  d = n & (DK - 1);
          ((_Float16*)Call)[((size_t)(b * NH + h) * DK + d) * SS + s] = (_Float16)v;
        } else if constexpr (MODE == MODE_SCORES) {
          float* C = (float*)Call + (size_t)z * SS * SS;
          float sv = v * scale;
          if (n > m) sv = -1e30f;          // causal mask
          C[(size_t)m * SS + n] = sv;
        } else { // MODE_PV: C = attn[b, s, h*64 + d]
          _Float16* C = (_Float16*)Call + (size_t)bb * SS * DM + hh * DK;
          C[(size_t)m * ldc + n] = (_Float16)v;
        }
      }
    }
}

__global__ __launch_bounds__(256) void cvt_f32_f16(const float* __restrict__ s,
                                                   _Float16* __restrict__ d, int n) {
  int i = (blockIdx.x * 256 + threadIdx.x) * 4;
  if (i < n) {
    v4f f = *(const v4f*)(s + i);
#pragma unroll
    for (int j = 0; j < 4; ++j) d[i + j] = (_Float16)f[j];
  }
}

// In-place row softmax over SS fp32 columns; one 256-thread block per row.
__global__ __launch_bounds__(256) void softmax_rows(float* __restrict__ p) {
  __shared__ float red[256];
  float* r = p + (size_t)blockIdx.x * SS;
  const int t = threadIdx.x;
  float m = -1e38f;
  for (int c = t; c < SS; c += 256) m = fmaxf(m, r[c]);
  red[t] = m; __syncthreads();
  for (int s = 128; s > 0; s >>= 1) { if (t < s) red[t] = fmaxf(red[t], red[t + s]); __syncthreads(); }
  m = red[0]; __syncthreads();
  float sum = 0.f;
  for (int c = t; c < SS; c += 256) { float e = __expf(r[c] - m); r[c] = e; sum += e; }
  red[t] = sum; __syncthreads();
  for (int s = 128; s > 0; s >>= 1) { if (t < s) red[t] += red[t + s]; __syncthreads(); }
  float inv = 1.f / red[0];
  for (int c = t; c < SS; c += 256) r[c] *= inv;
}

extern "C" void kernel_launch(void* const* d_in, const int* in_sizes, int n_in,
                              void* d_out, int out_size, void* d_ws, size_t ws_size,
                              hipStream_t stream) {
  const float* query = (const float*)d_in[0];
  const float* keyin = (const float*)d_in[1];
  const float* value = (const float*)d_in[2];
  // d_in[3] = mask (int32 tril) — causal mask is applied analytically.
  const float* Wq = (const float*)d_in[4];
  const float* Wk = (const float*)d_in[5];
  const float* Wv = (const float*)d_in[6];
  const float* Wo = (const float*)d_in[7];

  const size_t nX = (size_t)BB * SS * DM;  // 4,194,304
  const size_t nW = (size_t)DM * DM;       // 1,048,576

  // ---- workspace layout (64 MB total) ----
  char* w = (char*)d_ws;
  auto take = [&](size_t elems) { _Float16* p = (_Float16*)w; w += elems * sizeof(_Float16); return p; };
  _Float16* hq  = take(nX);   // query in f16
  _Float16* hk  = take(nX);   // key   in f16
  _Float16* hv  = take(nX);   // value in f16
  _Float16* fWq = take(nW);
  _Float16* fWk = take(nW);
  _Float16* fWv = take(nW);
  _Float16* fWo = take(nW);
  _Float16* Qp  = take(nX);   // Q proj  [B*S, DM]
  _Float16* Kp  = take(nX);   // K proj  [B*S, DM]
  _Float16* Vt  = take(nX);   // V proj transposed per head [B,H,DK,S]
  _Float16* At  = take(nX);   // attention context [B*S, DM]

  float* outp  = (float*)d_out;          // [B,S,DM]
  float* probs = (float*)d_out + nX;     // [B,H,S,S]

  // ---- fp32 -> f16 staging ----
  cvt_f32_f16<<<(int)(nX / 1024), 256, 0, stream>>>(query, hq, (int)nX);
  cvt_f32_f16<<<(int)(nX / 1024), 256, 0, stream>>>(keyin, hk, (int)nX);
  cvt_f32_f16<<<(int)(nX / 1024), 256, 0, stream>>>(value, hv, (int)nX);
  cvt_f32_f16<<<(int)(nW / 1024), 256, 0, stream>>>(Wq, fWq, (int)nW);
  cvt_f32_f16<<<(int)(nW / 1024), 256, 0, stream>>>(Wk, fWk, (int)nW);
  cvt_f32_f16<<<(int)(nW / 1024), 256, 0, stream>>>(Wv, fWv, (int)nW);
  cvt_f32_f16<<<(int)(nW / 1024), 256, 0, stream>>>(Wo, fWo, (int)nW);

  const int Mrows = BB * SS;  // 4096
  // ---- projections: X @ W^T ----
  gemm_wmma<128, 128, MODE_F16><<<dim3(DM / 128, Mrows / 128, 1), 256, 0, stream>>>(
      hq, DM, fWq, DM, Qp, DM, DM, 1.0f);
  gemm_wmma<128, 128, MODE_F16><<<dim3(DM / 128, Mrows / 128, 1), 256, 0, stream>>>(
      hk, DM, fWk, DM, Kp, DM, DM, 1.0f);
  gemm_wmma<128, 128, MODE_VT><<<dim3(DM / 128, Mrows / 128, 1), 256, 0, stream>>>(
      hv, DM, fWv, DM, Vt, DM, DM, 1.0f);

  // ---- scores = Q K^T / sqrt(DK) + causal mask  (into probs region) ----
  gemm_wmma<128, 128, MODE_SCORES><<<dim3(SS / 128, SS / 128, BB * NH), 256, 0, stream>>>(
      Qp, DM, Kp, DM, probs, SS, DK, 0.125f);

  // ---- softmax rows in place ----
  softmax_rows<<<BB * NH * SS, 256, 0, stream>>>(probs);

  // ---- context = probs @ V  (fp32 probs staged to f16 in LDS) ----
  gemm_wmma<128, 64, MODE_PV><<<dim3(1, SS / 128, BB * NH), 256, 0, stream>>>(
      probs, SS, Vt, SS, At, DM, SS, 1.0f);

  // ---- out = context @ Wo^T (fp32) ----
  gemm_wmma<128, 128, MODE_F32><<<dim3(DM / 128, Mrows / 128, 1), 256, 0, stream>>>(
      At, DM, fWo, DM, outp, DM, DM, 1.0f);
}
